// DiffusionDecoder_62062277427453
// MI455X (gfx1250) — matrix operand order repaired and checked by reading, not compiled, not run on hardware
//
#include <hip/hip_runtime.h>

// CDNA5 / gfx1250, wave32. D = E @ OneHot via v_wmma_f32_16x16x32_f16.

typedef __attribute__((ext_vector_type(16))) _Float16 v16h;
typedef __attribute__((ext_vector_type(8)))  float    v8f;

#define N_CELLS  16384
#define N_SPOTS  4096
#define N_LABELS 512
#define NT       8            // 16-wide label tiles per wave -> 128 labels/wave
#define NSUPER   4            // 512 / (16*NT)
#define CHUNKS   (N_SPOTS / 32)

// ---------------- kernel 1: zero label counts ----------------
__global__ void zero_counts_k(int* __restrict__ counts) {
    counts[threadIdx.x] = 0;
}

// ---------------- kernel 2: histogram of spot labels ----------------
__global__ void count_labels_k(const int* __restrict__ labels, int* __restrict__ counts) {
    int i = blockIdx.x * blockDim.x + threadIdx.x;
    if (i < N_SPOTS) atomicAdd(&counts[labels[i]], 1);
}

// ---------------- kernel 3: pack one-hot B into WMMA fragment order ----------
// Fragment idx = chunk*1024 + label*2 + half  (128*512*2 = 131072 fragments)
// Each fragment = 16 f16 = 32 bytes: B[k0 + half*16 + j][label], j = 0..15.
__global__ void build_b_k(const int* __restrict__ labels, _Float16* __restrict__ bh) {
    int idx = blockIdx.x * blockDim.x + threadIdx.x;     // 0 .. 131071
    int h     = idx & 1;
    int l     = (idx >> 1) & (N_LABELS - 1);
    int chunk = idx >> 10;
    int kbase = chunk * 32 + h * 16;
    v16h v;
#pragma unroll
    for (int j = 0; j < 16; ++j)
        v[j] = (labels[kbase + j] == l) ? (_Float16)1.0f : (_Float16)0.0f;
    *(v16h*)(bh + (size_t)idx * 16) = v;
}

// ---------------- kernel 4: fused exp + WMMA GEMM + epilogue ----------------
__global__ void __launch_bounds__(256)
diffusion_gemm_k(const float* __restrict__ z,
                 const float* __restrict__ dconst,
                 const float* __restrict__ ex,
                 const float* __restrict__ ey,
                 const int*   __restrict__ counts,
                 const _Float16* __restrict__ bh,
                 float* __restrict__ out) {
    const int lane   = threadIdx.x & 31;
    const int wv     = threadIdx.x >> 5;          // 8 waves / block
    const int lanelo = lane & 15;
    const int hi     = lane >> 4;                 // half-wave select
    const int n_super = blockIdx.x & (NSUPER - 1);      // same for all waves in block
    const int m_tile  = (blockIdx.x >> 2) * 8 + wv;     // 0 .. 1023
    const int n_base  = n_super * (16 * NT);

    const int   row = m_tile * 16 + lanelo;       // A row owned by this lane
    const float zx  = z[2 * row + 0];
    const float zy  = z[2 * row + 1];
    const float Dv  = dconst[0];
    // exp(-d^2/(2D)) = exp2(d^2 * c2), c2 = -1/(2 D ln2)
    const float c2  = -1.442695040888963f / (2.0f * Dv);

    const int kb = hi * 16;   // A-fragment K base offset (contiguous: K 16*hi .. 16*hi+15)
    const int hB = hi;        // B-fragment half (lanes<16: K 0..15, lanes>=16: K 16..31)

    v8f acc[NT] = {};

    for (int chunk = 0; chunk < CHUNKS; ++chunk) {
        const int k0 = chunk * 32;
        // A K indices per lane: [k0+kb, k0+kb+16)
        const float4 xa = *(const float4*)(ex + k0 + kb);
        const float4 xb = *(const float4*)(ex + k0 + kb + 4);
        const float4 xc = *(const float4*)(ex + k0 + kb + 8);
        const float4 xd = *(const float4*)(ex + k0 + kb + 12);
        const float4 ya = *(const float4*)(ey + k0 + kb);
        const float4 yb = *(const float4*)(ey + k0 + kb + 4);
        const float4 yc = *(const float4*)(ey + k0 + kb + 8);
        const float4 yd = *(const float4*)(ey + k0 + kb + 12);

        const float exv[16] = {xa.x, xa.y, xa.z, xa.w, xb.x, xb.y, xb.z, xb.w,
                               xc.x, xc.y, xc.z, xc.w, xd.x, xd.y, xd.z, xd.w};
        const float eyv[16] = {ya.x, ya.y, ya.z, ya.w, yb.x, yb.y, yb.z, yb.w,
                               yc.x, yc.y, yc.z, yc.w, yd.x, yd.y, yd.z, yd.w};

        v16h a;
#pragma unroll
        for (int i = 0; i < 16; ++i) {
            float dx = exv[i] - zx;
            float dy = eyv[i] - zy;
            float d2 = __builtin_fmaf(dx, dx, dy * dy);
            a[i] = (_Float16)__builtin_amdgcn_exp2f(d2 * c2);   // v_exp_f32
        }

        // B fragments for this chunk/half; label row stride = 32 halves
        const _Float16* bbase = bh + ((size_t)chunk * (N_LABELS * 2) + hB) * 16;
#pragma unroll
        for (int t = 0; t < NT; ++t) {
            const int l = n_base + t * 16 + lanelo;
            v16h b = *(const v16h*)(bbase + (size_t)l * 32);
            acc[t] = __builtin_amdgcn_wmma_f32_16x16x32_f16(
                         /*neg_a=*/false, a, /*neg_b=*/false, b,
                         /*c_mod=*/(short)0, acc[t],
                         /*reuse_a=*/false, /*reuse_b=*/false);
        }
    }

    // Epilogue: out = norm * acc + nu * count[label]
    const float norm    = 0.15915494309189535f / Dv;   // 1/(2*pi*D)
    const int   rowbase = m_tile * 16 + hi * 8;        // C/D: VGPR r -> row r (+8 upper half)
#pragma unroll
    for (int t = 0; t < NT; ++t) {
        const int   l   = n_base + t * 16 + lanelo;
        const float nuc = 1e-12f * (float)counts[l];
#pragma unroll
        for (int r = 0; r < 8; ++r) {
            out[(size_t)(rowbase + r) * N_LABELS + l] =
                __builtin_fmaf(norm, acc[t][r], nuc);
        }
    }
}

// ---------------- launcher ----------------
extern "C" void kernel_launch(void* const* d_in, const int* in_sizes, int n_in,
                              void* d_out, int out_size, void* d_ws, size_t ws_size,
                              hipStream_t stream) {
    const float* z      = (const float*)d_in[0];   // (16384, 2)
    const float* dconst = (const float*)d_in[1];   // scalar D
    const float* ex     = (const float*)d_in[2];   // (4096,)
    const float* ey     = (const float*)d_in[3];   // (4096,)
    const int*   labels = (const int*)d_in[4];     // (4096,) int32
    float*       out    = (float*)d_out;           // (16384, 512) f32

    // ws layout: [0, 2KB) label counts, [4KB, 4KB + 4MB) packed f16 one-hot B
    int*      counts = (int*)d_ws;
    _Float16* bh     = (_Float16*)((char*)d_ws + 4096);

    zero_counts_k <<<1,   N_LABELS, 0, stream>>>(counts);
    count_labels_k<<<16,  256,      0, stream>>>(labels, counts);
    build_b_k     <<<512, 256,      0, stream>>>(labels, bh);   // 131072 fragments
    diffusion_gemm_k<<<512, 256,    0, stream>>>(z, dconst, ex, ey, counts, bh, out);
}
